// VWModel_39694087750256
// MI455X (gfx1250) — matrix operand order repaired and checked by reading, compiled, not run on hardware
//
#include <hip/hip_runtime.h>

typedef __attribute__((ext_vector_type(16))) _Float16 v16h;
typedef __attribute__((ext_vector_type(8)))  _Float16 v8h;
typedef __attribute__((ext_vector_type(4)))  _Float16 v4h;
typedef __attribute__((ext_vector_type(8)))  float    v8f;

constexpr int B_  = 4096;
constexpr int T_  = 64;
constexpr int F_  = 32;
constexpr int U1_ = 256;
constexpr int U2_ = 128;

constexpr int BM       = 32;   // batch rows per workgroup (two 16-row M tiles)
constexpr int RT       = BM / 16;
constexpr int NTHREADS = 256;  // 8 waves (wave32)
constexpr int NWAVES   = 8;

// ---------------------------------------------------------------------------
// Weight packing: fp32 [K,N] row-major -> f16 WMMA B-fragments.
// Fragment (kt,nt) covers K rows [kt*32, kt*32+32), N cols [nt*16, nt*16+16).
// Per ISA 16-bit B layout (32x16): lane l holds N = l&15; lanes 0-15 hold
// K=0..7,16..23, lanes 16-31 hold K=8..15,24..31. Flattened element index
// within a fragment = lane*16 + j, where K = (lane>>4)*8 + (j&7) + (j>>3)*16, N = lane&15.
// Fragment order: frag = nt*KT + kt  (kt-contiguous for the GEMM K-loop).
// ---------------------------------------------------------------------------
__global__ __launch_bounds__(NTHREADS) void pack_weight_kernel(
    const float* __restrict__ src, _Float16* __restrict__ dst, int K, int N) {
  const int KT    = K >> 5;
  const int total = (K >> 5) * (N >> 4) * 512;
  int idx = blockIdx.x * NTHREADS + threadIdx.x;
  if (idx >= total) return;
  int e    = idx & 511;
  int frag = idx >> 9;
  int kt   = frag % KT;
  int nt   = frag / KT;
  int lane = e >> 4;
  int j    = e & 15;
  int n = nt * 16 + (lane & 15);
  int k = kt * 32 + (lane >> 4) * 8 + (j & 7) + (j >> 3) * 16;
  dst[idx] = (_Float16)src[(size_t)k * N + n];
}

// ---------------------------------------------------------------------------
// A-fragment load from a row-major f16 [16 x stride] tile at K offset kofs.
// ISA 16-bit A layout (16x32): lane l -> M = l&15; K chunks:
// lanes 0-15:  VGPR0-3 = K 0..7,  VGPR4-7 = K 16..23
// lanes 16-31: VGPR0-3 = K 8..15, VGPR4-7 = K 24..31
// => two contiguous 8-half (16B) loads per lane at K = base and base+16.
// ---------------------------------------------------------------------------
__device__ inline v16h load_a_frag(const _Float16* base, int stride, int kofs,
                                   int lane) {
  const int m  = lane & 15;
  const int kb = kofs + (lane >> 4) * 8;
  const _Float16* p = base + m * stride + kb;
  v8h lo = *(const v8h*)(p);
  v8h hi = *(const v8h*)(p + 16);
  return __builtin_shufflevector(lo, hi, 0, 1, 2, 3, 4, 5, 6, 7, 8, 9, 10, 11,
                                 12, 13, 14, 15);
}

__device__ inline v8f splat8(float v) {
  return (v8f){v, v, v, v, v, v, v, v};
}

// ---------------------------------------------------------------------------
// One GRU layer over the full sequence for a BM-row batch tile.
//   gates = sigmoid([x,h] Wg + bg); r,u = split(gates)
//   c     = tanh([x, r*h] Wc + bc)
//   h'    = u*h + (1-u)*c
// GEMMs run as v_wmma_f32_16x16x32_f16 (fp32 accumulate). Each wave owns a
// set of N tiles and computes BOTH 16-row M tiles per B-fragment load (2x
// B reuse). Candidate weights are register-resident across all timesteps.
// Gates tile loop intentionally NOT unrolled: full unroll makes the compiler
// hoist all gates B-fragments into VGPRs and spill to scratch.
// ---------------------------------------------------------------------------
template <int XD, int U, bool X_F32, bool WRITE_SEQ, bool DO_DENSE>
__global__ __launch_bounds__(NTHREADS) void gru_layer_kernel(
    const void* __restrict__ xin,          // [B,T,XD] f32 (layer1) or f16
    const _Float16* __restrict__ wg_pk,    // packed gates weights [XD+U, 2U]
    const float* __restrict__ bg,          // [2U]
    const _Float16* __restrict__ wc_pk,    // packed cand weights [XD+U, U]
    const float* __restrict__ bc,          // [U]
    _Float16* __restrict__ hseq,           // [B,T,U] f16 (if WRITE_SEQ)
    const float* __restrict__ Wd,          // [U,F] (if DO_DENSE)
    const float* __restrict__ bd,          // [F]
    float* __restrict__ out)               // [B,F] (if DO_DENSE)
{
  constexpr int KTX = XD / 32;
  constexpr int KTH = U / 32;
  constexpr int KT  = KTX + KTH;
  constexpr int NTG = (2 * U) / 16;        // gate col tiles
  constexpr int NTC = U / 16;              // cand col tiles
  constexpr int GPW = NTG / NWAVES;        // gate tiles per wave
  constexpr int CPW = NTC / NWAVES;        // cand tiles per wave

  // dynamic LDS carve-up (avoids static-LDS size cap; hw allows 320KB/WG)
  extern __shared__ char smem[];
  _Float16* xs   = (_Float16*)smem;        // [BM][XD]  x_t, f16
  _Float16* hs16 = xs + BM * XD;           // [BM][U]   h, f16 (A-frags)
  _Float16* rhs  = hs16 + BM * U;          // [BM][U]   r*h, f16 (A-frags)
  float*    hs32 = (float*)(rhs + BM * U); // [BM][U]   h, fp32 (state)
  float*    us   = hs32 + BM * U;          // [BM][U]   update gate u, fp32

  const int tid   = threadIdx.x;
  const int lane  = tid & 31;
  const int wave  = tid >> 5;
  const int row0  = blockIdx.x * BM;
  const int nlo   = lane & 15;        // N within a C/D tile
  const int mbase = (lane >> 4) * 8;  // M base within a C/D tile

  for (int i = tid; i < BM * U; i += NTHREADS) {
    hs32[i] = 0.f;
    hs16[i] = (_Float16)0.f;
  }

  // ---- per-lane bias preload (hoisted out of the time loop) ----
  float bgv[GPW], bcv[CPW];
#pragma unroll
  for (int g = 0; g < GPW; ++g) bgv[g] = bg[(wave * GPW + g) * 16 + nlo];
#pragma unroll
  for (int g = 0; g < CPW; ++g) bcv[g] = bc[(wave * CPW + g) * 16 + nlo];

  // ---- candidate weights register-resident across all timesteps ----
  v16h wcf[CPW][KT];
#pragma unroll
  for (int g = 0; g < CPW; ++g)
#pragma unroll
    for (int kt = 0; kt < KT; ++kt)
      wcf[g][kt] = *(const v16h*)(
          wc_pk + ((size_t)((wave * CPW + g) * KT + kt)) * 512 + lane * 16);

  __syncthreads();

  for (int t = 0; t < T_; ++t) {
    // ---- stage x_t into LDS (f16), vectorized ----
    if constexpr (X_F32) {
      const float* xp = (const float*)xin;
      constexpr int XC = XD / 4;  // float4 chunks per row
      for (int i = tid; i < BM * XC; i += NTHREADS) {
        int r = i / XC, c = i % XC;
        const float4 v =
            *(const float4*)(xp + (size_t)(row0 + r) * (T_ * XD) + t * XD +
                             c * 4);
        v4h h4 = {(_Float16)v.x, (_Float16)v.y, (_Float16)v.z, (_Float16)v.w};
        *(v4h*)(xs + r * XD + c * 4) = h4;
      }
    } else {
      const _Float16* xp = (const _Float16*)xin;
      constexpr int XC = XD / 8;  // 16-byte chunks per row
      for (int i = tid; i < BM * XC; i += NTHREADS) {
        int r = i / XC, c = i % XC;
        *(v8h*)(xs + r * XD + c * 8) =
            *(const v8h*)(xp + (size_t)(row0 + r) * (T_ * XD) + t * XD +
                          c * 8);
      }
    }
    __syncthreads();

    // ---- gates GEMM: [BM, XD+U] x [XD+U, 2U]; 2 M-tiles per B-frag ----
    for (int g = 0; g < GPW; ++g) {
      const int nt   = wave * GPW + g;
      const int ncol = nt * 16 + nlo;
      v8f acc0 = splat8(bgv[g]);
      v8f acc1 = splat8(bgv[g]);
#pragma unroll
      for (int kt = 0; kt < KT; ++kt) {
        const _Float16* abase = (kt < KTX) ? xs : hs16;
        const int astride     = (kt < KTX) ? XD : U;
        const int akofs       = (kt < KTX) ? kt * 32 : (kt - KTX) * 32;
        v16h a0 = load_a_frag(abase, astride, akofs, lane);
        v16h a1 = load_a_frag(abase + 16 * astride, astride, akofs, lane);
        v16h b  = *(const v16h*)(wg_pk + ((size_t)(nt * KT + kt)) * 512 +
                                 lane * 16);
        acc0 = __builtin_amdgcn_wmma_f32_16x16x32_f16(false, a0, false, b,
                                                      (short)0, acc0, false,
                                                      false);
        acc1 = __builtin_amdgcn_wmma_f32_16x16x32_f16(false, a1, false, b,
                                                      (short)0, acc1, false,
                                                      false);
      }
      // sigmoid + scatter (wave-uniform r/u split: tile never straddles U)
      if (ncol < U) {
#pragma unroll
        for (int j = 0; j < 8; ++j) {
          int m0 = mbase + j, m1 = 16 + mbase + j;
          float g0 = 1.f / (1.f + __expf(-acc0[j]));
          float g1 = 1.f / (1.f + __expf(-acc1[j]));
          rhs[m0 * U + ncol] = (_Float16)(g0 * hs32[m0 * U + ncol]);
          rhs[m1 * U + ncol] = (_Float16)(g1 * hs32[m1 * U + ncol]);
        }
      } else {
#pragma unroll
        for (int j = 0; j < 8; ++j) {
          int m0 = mbase + j, m1 = 16 + mbase + j;
          us[m0 * U + ncol - U] = 1.f / (1.f + __expf(-acc0[j]));
          us[m1 * U + ncol - U] = 1.f / (1.f + __expf(-acc1[j]));
        }
      }
    }
    __syncthreads();

    // ---- candidate GEMM: [BM, XD+U] x [XD+U, U], then state update ----
#pragma unroll
    for (int g = 0; g < CPW; ++g) {
      const int nt   = wave * CPW + g;
      const int ncol = nt * 16 + nlo;
      v8f acc0 = splat8(bcv[g]);
      v8f acc1 = splat8(bcv[g]);
#pragma unroll
      for (int kt = 0; kt < KT; ++kt) {
        const _Float16* abase = (kt < KTX) ? xs : rhs;
        const int astride     = (kt < KTX) ? XD : U;
        const int akofs       = (kt < KTX) ? kt * 32 : (kt - KTX) * 32;
        v16h a0 = load_a_frag(abase, astride, akofs, lane);
        v16h a1 = load_a_frag(abase + 16 * astride, astride, akofs, lane);
        v16h b  = wcf[g][kt];  // register-resident
        acc0 = __builtin_amdgcn_wmma_f32_16x16x32_f16(false, a0, false, b,
                                                      (short)0, acc0, false,
                                                      false);
        acc1 = __builtin_amdgcn_wmma_f32_16x16x32_f16(false, a1, false, b,
                                                      (short)0, acc1, false,
                                                      false);
      }
#pragma unroll
      for (int j = 0; j < 8; ++j) {
#pragma unroll
        for (int r2 = 0; r2 < RT; ++r2) {
          int m = r2 * 16 + mbase + j;
          float c  = tanhf(r2 ? acc1[j] : acc0[j]);
          float uu = us[m * U + ncol];
          float hn = uu * hs32[m * U + ncol] + (1.f - uu) * c;
          hs32[m * U + ncol] = hn;
          hs16[m * U + ncol] = (_Float16)hn;
          if constexpr (WRITE_SEQ)
            hseq[(size_t)(row0 + m) * (T_ * U) + t * U + ncol] = (_Float16)hn;
        }
      }
    }
    __syncthreads();
  }

  // ---- dense head on the final hidden state ----
  if constexpr (DO_DENSE) {
    for (int i = tid; i < BM * F_; i += NTHREADS) {
      int r = i / F_, f = i % F_;
      float s = bd[f];
      for (int k = 0; k < U; ++k) s += hs32[r * U + k] * Wd[k * F_ + f];
      out[(size_t)(row0 + r) * F_ + f] = s;
    }
  }
}

// ---------------------------------------------------------------------------
extern "C" void kernel_launch(void* const* d_in, const int* in_sizes, int n_in,
                              void* d_out, int out_size, void* d_ws,
                              size_t ws_size, hipStream_t stream) {
  (void)in_sizes; (void)n_in; (void)out_size; (void)ws_size;

  const float* frames = (const float*)d_in[0];
  const float* W1g = (const float*)d_in[1];
  const float* b1g = (const float*)d_in[2];
  const float* W1c = (const float*)d_in[3];
  const float* b1c = (const float*)d_in[4];
  const float* W2g = (const float*)d_in[5];
  const float* b2g = (const float*)d_in[6];
  const float* W2c = (const float*)d_in[7];
  const float* b2c = (const float*)d_in[8];
  const float* Wd  = (const float*)d_in[9];
  const float* bd  = (const float*)d_in[10];

  // workspace carve-up (f16 elements)
  _Float16* ws = (_Float16*)d_ws;
  size_t o = 0;
  _Float16* w1g_pk = ws + o; o += (size_t)(288 / 32) * (512 / 16) * 512;
  _Float16* w1c_pk = ws + o; o += (size_t)(288 / 32) * (256 / 16) * 512;
  _Float16* w2g_pk = ws + o; o += (size_t)(384 / 32) * (256 / 16) * 512;
  _Float16* w2c_pk = ws + o; o += (size_t)(384 / 32) * (128 / 16) * 512;
  _Float16* h1seq  = ws + o; o += (size_t)B_ * T_ * U1_;

  auto pack = [&](const float* src, _Float16* dst, int K, int N) {
    int total = (K / 32) * (N / 16) * 512;
    pack_weight_kernel<<<(total + NTHREADS - 1) / NTHREADS, NTHREADS, 0,
                         stream>>>(src, dst, K, N);
  };
  pack(W1g, w1g_pk, F_ + U1_, 2 * U1_);   // 288 x 512
  pack(W1c, w1c_pk, F_ + U1_, U1_);       // 288 x 256
  pack(W2g, w2g_pk, U1_ + U2_, 2 * U2_);  // 384 x 256
  pack(W2c, w2c_pk, U1_ + U2_, U2_);      // 384 x 128

  dim3 grid(B_ / BM);
  // dynamic LDS: xs + hs16 + rhs (f16) + hs32 + us (f32)
  size_t smem1 = (size_t)BM * F_ * 2 + 2 * (size_t)BM * U1_ * 2 +
                 2 * (size_t)BM * U1_ * 4;                       // ~100 KB
  size_t smem2 = (size_t)BM * U1_ * 2 + 2 * (size_t)BM * U2_ * 2 +
                 2 * (size_t)BM * U2_ * 4;                       // 64 KB
  // layer 1: fp32 frames in, f16 h1 sequence out
  gru_layer_kernel<F_, U1_, true, true, false><<<grid, NTHREADS, smem1,
                                                 stream>>>(
      frames, w1g_pk, b1g, w1c_pk, b1c, h1seq, nullptr, nullptr, nullptr);
  // layer 2: f16 h1 sequence in, dense head to d_out
  gru_layer_kernel<U1_, U2_, false, false, true><<<grid, NTHREADS, smem2,
                                                   stream>>>(
      h1seq, w2g_pk, b2g, w2c_pk, b2c, nullptr, Wd, bd, (float*)d_out);
}
